// MaskedGINE_1511828488905
// MI455X (gfx1250) — compile-verified, matrix operand order checked by reading, not run on hardware
//
#include <hip/hip_runtime.h>
#include <hip/hip_bf16.h>

// ---------------------------------------------------------------------------
// MaskedGINE on MI455X (gfx1250).
//  - bf16 WMMA GEMMs (v_wmma_f32_16x16x32_bf16), f32 accumulate
//  - double-buffered LDS tiles, register staging, 1 barrier / k-step
//  - b128 global loads on the in-bounds fast path (A); coalesced column
//    loads + packed-pair LDS stores (B)
//  - native __bf16 casts so the backend can select packed bf16 converts
//    (replaces the manual RNE bit-twiddling)
//  - templated prologue modes: 0 plain(+residual), 1 node-attr MLP,
//    2 edge-attr MLP; epilogue: bias (+ atom_emb[z] gather) (+relu), f32/bf16
//  - message/scatter: block-per-edge, global_atomic_add_f32
//  - BatchNorm: register column sums -> global atomic partials -> fused apply
// Memory-bound (~7 GB HBM traffic -> ~300 us floor @ 23.3 TB/s).
// ---------------------------------------------------------------------------

typedef __attribute__((ext_vector_type(16))) __bf16 v16bf;
typedef __attribute__((ext_vector_type(16))) unsigned short v16us;
typedef __attribute__((ext_vector_type(8))) float v8f;
typedef __attribute__((ext_vector_type(4))) float f4;
typedef __attribute__((ext_vector_type(2))) __bf16 v2bf;

union Frag16 { v16us us; v16bf bf; };

__device__ __forceinline__ unsigned short f32_to_bf16_bits(float f) {
  __bf16 b = (__bf16)f;                       // native cvt, RNE
  return __builtin_bit_cast(unsigned short, b);
}
__device__ __forceinline__ unsigned pack_bf16x2(float lo, float hi) {
  v2bf p;
  p[0] = (__bf16)lo;                          // packed pair -> v_cvt_pk-able
  p[1] = (__bf16)hi;
  return __builtin_bit_cast(unsigned, p);
}
__device__ __forceinline__ float bf16_bits_to_f32(unsigned short s) {
  unsigned u = ((unsigned)s) << 16;
  return __builtin_bit_cast(float, u);
}

#define BM 64
#define BN 64
#define BK 32
#define LSTR 34   // ushort stride: even (32-bit LDS accesses), conflict-friendly

// One k-step of WMMA from an LDS tile pair (A fragment layout per ISA 7.12.2).
__device__ __forceinline__ void wmma_step(
    const unsigned short* __restrict__ As, const unsigned short* __restrict__ Bs,
    int wm, int wn, int half, int l16, v8f& acc0, v8f& acc1) {
  Frag16 af;
  {
    const unsigned* arow = (const unsigned*)&As[(16 * wm + l16) * LSTR];
    #pragma unroll
    for (int v = 0; v < 8; ++v) {
      int kb = (v < 4) ? (2 * v + 8 * half) : (16 + 2 * (v - 4) + 8 * half);
      unsigned p = arow[kb >> 1];
      af.us[2 * v]     = (unsigned short)(p & 0xFFFFu);
      af.us[2 * v + 1] = (unsigned short)(p >> 16);
    }
  }
  #pragma unroll
  for (int t = 0; t < 2; ++t) {
    Frag16 bfr;
    const unsigned* brow = (const unsigned*)&Bs[(32 * wn + 16 * t + l16) * LSTR];
    #pragma unroll
    for (int v = 0; v < 8; ++v) {
      int kb = (v < 4) ? (2 * v + 8 * half) : (16 + 2 * (v - 4) + 8 * half);
      unsigned p = brow[kb >> 1];
      bfr.us[2 * v]     = (unsigned short)(p & 0xFFFFu);
      bfr.us[2 * v + 1] = (unsigned short)(p >> 16);
    }
    if (t == 0)
      acc0 = __builtin_amdgcn_wmma_f32_16x16x32_bf16(
          false, af.bf, false, bfr.bf, (short)0, acc0, false, false);
    else
      acc1 = __builtin_amdgcn_wmma_f32_16x16x32_bf16(
          false, af.bf, false, bfr.bf, (short)0, acc1, false, false);
  }
}

template <int AMODE, int OUTBF16>
__global__ __launch_bounds__(256) void gemm_wmma_bf16_kernel(
    const float* __restrict__ A,  const float* __restrict__ A2,
    const float* __restrict__ f0, const float* __restrict__ f1,
    const float* __restrict__ aw, const float* __restrict__ ab,
    const float* __restrict__ W,  const float* __restrict__ bias,
    const float* __restrict__ gtab, const int* __restrict__ gidx,
    void* __restrict__ outp,
    int M, int K, int Nout, int ldOut, int do_relu)
{
  __shared__ unsigned short As[2][BM * LSTR];
  __shared__ unsigned short Bs[2][BN * LSTR];

  const int tid  = threadIdx.x;
  const int lane = tid & 31;
  const int wave = tid >> 5;
  const int wm   = wave & 3;
  const int wn   = wave >> 2;
  const int half = lane >> 4;
  const int l16  = lane & 15;

  const int row0 = blockIdx.x * BM;
  const int col0 = blockIdx.y * BN;

  // ---- tile-load thread mapping ----
  const int rA = tid >> 2;            // 0..63  A row
  const int cA = (tid & 3) * 8;       // 0,8,16,24
  const int nB = tid & 63;            // 0..63  B col
  const int kB = (tid >> 6) * 8;      // 0,8,16,24 (8 consecutive k)
  const int aRow = row0 + rA;
  const int bCol = col0 + nB;
  const bool aRowOK = aRow < M;
  const bool bColOK = bCol < Nout;
  const bool aVec = aRowOK && ((K & 3) == 0);   // b128 path legal

  // Hoisted per-row scalars for the fused stage-1 MLP prologues.
  float s0 = 0.f, s1 = 0.f, s2 = 0.f;
  if (AMODE == 1 && aRowOK) { s0 = f0[aRow]; s1 = f1[aRow]; }
  if (AMODE == 2 && aRowOK) {
    const float* ea = A + (long)aRow * 3;
    s0 = ea[0]; s1 = ea[1]; s2 = ea[2];
  }

  auto loadA = [&](int k0, float* va) {
    if (AMODE == 0) {
      const long base = (long)aRow * K + k0 + cA;
      if (aVec && (k0 + BK <= K)) {
        f4 x = *(const f4*)(A + base);
        f4 y = *(const f4*)(A + base + 4);
        va[0]=x.x; va[1]=x.y; va[2]=x.z; va[3]=x.w;
        va[4]=y.x; va[5]=y.y; va[6]=y.z; va[7]=y.w;
        if (A2) {
          f4 p = *(const f4*)(A2 + base);
          f4 q = *(const f4*)(A2 + base + 4);
          va[0]+=p.x; va[1]+=p.y; va[2]+=p.z; va[3]+=p.w;
          va[4]+=q.x; va[5]+=q.y; va[6]+=q.z; va[7]+=q.w;
        }
      } else {
        #pragma unroll
        for (int j = 0; j < 8; ++j) {
          int k = k0 + cA + j;
          float v = 0.f;
          if (aRowOK && k < K) {
            v = A[(long)aRow * K + k];
            if (A2) v += A2[(long)aRow * K + k];
          }
          va[j] = v;
        }
      }
    } else {
      #pragma unroll
      for (int j = 0; j < 8; ++j) {
        int k = k0 + cA + j;
        float v = 0.f;
        if (aRowOK && k < K) {
          if (AMODE == 1) v = s0 * aw[k] + s1 * aw[K + k] + ab[k];
          else            v = s0 * aw[k] + s1 * aw[K + k] + s2 * aw[2 * K + k] + ab[k];
          v = v > 0.f ? v : 0.f;
        }
        va[j] = v;
      }
    }
  };

  // Each thread loads one column (coalesced across lanes), 8 consecutive k.
  auto loadB = [&](int k0, float* vb) {
    #pragma unroll
    for (int j = 0; j < 8; ++j) {
      int k = k0 + kB + j;
      vb[j] = (bColOK && k < K) ? W[(long)k * Nout + bCol] : 0.f;
    }
  };

  auto storeTiles = [&](unsigned short* as, unsigned short* bs,
                        const float* va, const float* vb) {
    unsigned* ad = (unsigned*)&as[rA * LSTR + cA];     // even ushort index
    #pragma unroll
    for (int j = 0; j < 4; ++j) ad[j] = pack_bf16x2(va[2 * j], va[2 * j + 1]);
    unsigned* bd = (unsigned*)&bs[nB * LSTR + kB];     // even ushort index
    #pragma unroll
    for (int j = 0; j < 4; ++j) bd[j] = pack_bf16x2(vb[2 * j], vb[2 * j + 1]);
  };

  v8f acc0 = {};
  v8f acc1 = {};

  const int ktiles = (K + BK - 1) / BK;
  float va[8], vb[8];
  loadA(0, va);
  loadB(0, vb);
  storeTiles(As[0], Bs[0], va, vb);
  __syncthreads();

  for (int kt = 0; kt < ktiles; ++kt) {
    const int p = kt & 1;
    const bool more = (kt + 1) < ktiles;
    if (more) {                        // issue next tile's global loads first
      loadA((kt + 1) * BK, va);
      loadB((kt + 1) * BK, vb);
    }
    wmma_step(As[p], Bs[p], wm, wn, half, l16, acc0, acc1);
    if (more) storeTiles(As[1 - p], Bs[1 - p], va, vb);
    __syncthreads();
  }

  // ---- epilogue: C/D layout VGPR r <-> M = r + 8*half ----
  #pragma unroll
  for (int t = 0; t < 2; ++t) {
    const v8f acc = t ? acc1 : acc0;
    int col = col0 + 32 * wn + 16 * t + l16;
    if (col >= Nout) continue;
    #pragma unroll
    for (int r = 0; r < 8; ++r) {
      int row = row0 + 16 * wm + 8 * half + r;
      if (row >= M) continue;
      float v = acc[r] + bias[col];
      if (gtab) v += gtab[(long)gidx[row] * Nout + col];
      if (do_relu) v = v > 0.f ? v : 0.f;
      if (OUTBF16)
        ((unsigned short*)outp)[(long)row * ldOut + col] = f32_to_bf16_bits(v);
      else
        ((float*)outp)[(long)row * ldOut + col] = v;
    }
  }
}

__global__ void zero_kernel(float* __restrict__ p, long n) {
  long i = (long)blockIdx.x * blockDim.x + threadIdx.x;
  if (i < n) p[i] = 0.f;
}

// Block per edge: msg = relu(h[src] + e); agg[dst] += msg (global f32 atomics).
__global__ void msg_scatter_kernel(const float* __restrict__ h,
                                   const unsigned short* __restrict__ ebf,
                                   const int* __restrict__ ei,
                                   float* __restrict__ agg, int E, int D) {
  int e = blockIdx.x;
  int d = threadIdx.x;
  if (d >= D) return;
  int s = ei[e];
  int t = ei[E + e];
  float v = h[(long)s * D + d] + bf16_bits_to_f32(ebf[(long)e * D + d]);
  v = v > 0.f ? v : 0.f;
  atomicAdd(&agg[(long)t * D + d], v);
}

// Block per 256 rows: register column sums -> one global atomic per column.
__global__ void bn_stats_kernel(const float* __restrict__ hh,
                                float* __restrict__ stats, int Nrows, int D) {
  int d = threadIdx.x;
  if (d >= D) return;
  long r0 = (long)blockIdx.x * 256;
  long r1 = r0 + 256; if (r1 > Nrows) r1 = Nrows;
  float s = 0.f, q = 0.f;
  for (long r = r0; r < r1; ++r) {
    float v = hh[r * D + d];
    s += v; q += v * v;
  }
  atomicAdd(&stats[d], s);
  atomicAdd(&stats[D + d], q);
}

// Block per row: h = relu((hh - mu) * rsqrt(var + eps) * gamma + beta).
__global__ void bn_apply_kernel(const float* __restrict__ hh,
                                const float* __restrict__ stats,
                                const float* __restrict__ gamma,
                                const float* __restrict__ beta,
                                float* __restrict__ hout, int Nrows, int D) {
  int d = threadIdx.x;
  if (d >= D) return;
  long i = (long)blockIdx.x * D + d;
  float invn = 1.f / (float)Nrows;
  float mu  = stats[d] * invn;
  float var = stats[D + d] * invn - mu * mu;
  float v = (hh[i] - mu) * rsqrtf(var + 1e-5f) * gamma[d] + beta[d];
  hout[i] = v > 0.f ? v : 0.f;
}

extern "C" void kernel_launch(void* const* d_in, const int* in_sizes, int n_in,
                              void* d_out, int out_size, void* d_ws, size_t ws_size,
                              hipStream_t stream) {
  const int*   z        = (const int*)  d_in[0];
  const float* chir     = (const float*)d_in[1];
  const float* fchg     = (const float*)d_in[2];
  const int*   ei       = (const int*)  d_in[3];
  const float* eattr    = (const float*)d_in[4];
  const float* atom_emb = (const float*)d_in[5];
  const float* nap_w1   = (const float*)d_in[6];
  const float* nap_b1   = (const float*)d_in[7];
  const float* nap_w2   = (const float*)d_in[8];
  const float* nap_b2   = (const float*)d_in[9];
  const float* ee_w1    = (const float*)d_in[10];
  const float* ee_b1    = (const float*)d_in[11];
  const float* ee_w2    = (const float*)d_in[12];
  const float* ee_b2    = (const float*)d_in[13];
  const float* gnn_w1   = (const float*)d_in[14];
  const float* gnn_b1   = (const float*)d_in[15];
  const float* gnn_w2   = (const float*)d_in[16];
  const float* gnn_b2   = (const float*)d_in[17];
  const float* bn_gamma = (const float*)d_in[18];
  const float* bn_beta  = (const float*)d_in[19];
  const float* ah_w     = (const float*)d_in[20];
  const float* ah_b     = (const float*)d_in[21];
  const float* ch_w     = (const float*)d_in[22];
  const float* ch_b     = (const float*)d_in[23];

  const int N  = in_sizes[0];
  const int E  = in_sizes[3] / 2;
  const int D  = in_sizes[7];
  const int V  = in_sizes[21];
  const int Kd = in_sizes[23];
  const int L  = in_sizes[15] / D;

  const long ND = (long)N * D;
  const long ED = (long)E * D;

  // Workspace (~600 MB @ reference sizes): h,hh,tmp,agg f32; e bf16; stats f32
  char* w = (char*)d_ws;
  float* h   = (float*)w;                     w += ND * sizeof(float);
  float* hh  = (float*)w;                     w += ND * sizeof(float);
  float* tmp = (float*)w;                     w += ND * sizeof(float);
  float* agg = (float*)w;                     w += ND * sizeof(float);
  unsigned short* ebuf = (unsigned short*)w;  w += ED * sizeof(unsigned short);
  float* stats = (float*)w;

  dim3 blk(256);
  const int dThreads = ((D + 63) / 64) * 64;   // 320 for D=300
  auto gx = [](long n) { return (unsigned)((n + 255) / 256); };
  dim3 gN((N + BM - 1) / BM, (D + BN - 1) / BN);
  dim3 gE((E + BM - 1) / BM, (D + BN - 1) / BN);

  // 1) x = atom_emb[z] + relu(na @ nap_w1 + b1) @ nap_w2 + b2
  gemm_wmma_bf16_kernel<1, 0><<<gN, blk, 0, stream>>>(
      nullptr, nullptr, chir, fchg, nap_w1, nap_b1,
      nap_w2, nap_b2, atom_emb, z, h, N, D, D, D, 0);

  // 2) e = relu(edge_attr @ ee_w1 + b1) @ ee_w2 + b2   (bf16 out)
  gemm_wmma_bf16_kernel<2, 1><<<gE, blk, 0, stream>>>(
      eattr, nullptr, nullptr, nullptr, ee_w1, ee_b1,
      ee_w2, ee_b2, nullptr, nullptr, ebuf, E, D, D, D, 0);

  for (int l = 0; l < L; ++l) {
    zero_kernel<<<gx(ND), blk, 0, stream>>>(agg, ND);
    msg_scatter_kernel<<<E, dThreads, 0, stream>>>(h, ebuf, ei, agg, E, D);

    // tmp = relu((h + agg) @ w1 + b1)   (residual fused into A-tile load)
    gemm_wmma_bf16_kernel<0, 0><<<gN, blk, 0, stream>>>(
        h, agg, nullptr, nullptr, nullptr, nullptr,
        gnn_w1 + (long)l * D * D, gnn_b1 + (long)l * D,
        nullptr, nullptr, tmp, N, D, D, D, /*relu=*/1);

    // hh = tmp @ w2 + b2
    gemm_wmma_bf16_kernel<0, 0><<<gN, blk, 0, stream>>>(
        tmp, nullptr, nullptr, nullptr, nullptr, nullptr,
        gnn_w2 + (long)l * D * D, gnn_b2 + (long)l * D,
        nullptr, nullptr, hh, N, D, D, D, 0);

    zero_kernel<<<gx(2 * (long)D), blk, 0, stream>>>(stats, 2 * (long)D);
    bn_stats_kernel<<<(N + 255) / 256, dThreads, 0, stream>>>(hh, stats, N, D);
    bn_apply_kernel<<<N, dThreads, 0, stream>>>(
        hh, stats, bn_gamma + (long)l * D, bn_beta + (long)l * D, h, N, D);
  }

  // Heads: logits [N,V] then dists [N,Kd], concatenated flat in d_out.
  float* logits = (float*)d_out;
  float* dists  = logits + (long)N * V;
  dim3 gH1((N + BM - 1) / BM, (V + BN - 1) / BN);
  gemm_wmma_bf16_kernel<0, 0><<<gH1, blk, 0, stream>>>(
      h, nullptr, nullptr, nullptr, nullptr, nullptr,
      ah_w, ah_b, nullptr, nullptr, logits, N, D, V, V, 0);
  dim3 gH2((N + BM - 1) / BM, (Kd + BN - 1) / BN);
  gemm_wmma_bf16_kernel<0, 0><<<gH2, blk, 0, stream>>>(
      h, nullptr, nullptr, nullptr, nullptr, nullptr,
      ch_w, ch_b, nullptr, nullptr, dists, N, D, Kd, Kd, 0);
}
